// NeuralVAEEncoder_67095979098219
// MI455X (gfx1250) — matrix operand order repaired and checked by reading, not compiled, not run on hardware
//
#include <hip/hip_runtime.h>
#include <hip/hip_bf16.h>

#define T_STEPS 1000
#define B_SZ    256
#define N_IN    256
#define H_SZ    128
#define G_SZ    384   // 3*H
#define BT      16    // batch rows per workgroup
#define L_SZ    16

typedef __attribute__((ext_vector_type(16))) __bf16 v16bf;
typedef __attribute__((ext_vector_type(8)))  float  v8f;

union FragB {
    uint4 q[2];
    v16bf v;
};

__device__ __forceinline__ unsigned short f2bf(float f) {
    unsigned u = __builtin_bit_cast(unsigned, f);
    u += 0x7FFFu + ((u >> 16) & 1u);   // round-to-nearest-even
    return (unsigned short)(u >> 16);
}

__device__ __forceinline__ unsigned pack2bf(float lo, float hi) {
    return (unsigned)f2bf(lo) | ((unsigned)f2bf(hi) << 16);
}

__device__ __forceinline__ v8f wmma_bf16(v16bf a, v16bf b, v8f c) {
    return __builtin_amdgcn_wmma_f32_16x16x32_bf16(
        /*neg_a=*/false, a, /*neg_b=*/false, b,
        /*c_mod=*/(short)0, c, /*reuse_a=*/false, /*reuse_b=*/false);
}

__device__ __forceinline__ void load_frag(FragB& f, const unsigned short* p0,
                                          const unsigned short* p1) {
    f.q[0] = *(const uint4*)p0;
    f.q[1] = *(const uint4*)p1;
}

// Dynamic LDS layout (in ushorts):
//   sWih : G_SZ*N_IN      (98304)   row-major [g][k] bf16
//   sWhh : G_SZ*H_SZ      (49152)   row-major [g][k] bf16
//   sX   : 2*BT*N_IN      ( 8192)   double-buffered x tile [b][k] bf16
//   sH   : 2*BT*H_SZ      ( 4096)   double-buffered hidden tile [b][j] bf16
// Total = 159,744 ushorts = 319,488 bytes < 320 KB/WGP.

__global__ void __launch_bounds__(256)
gru_dir_kernel(const float* __restrict__ y_aux,
               const float* __restrict__ Wih_f, const float* __restrict__ Whh_f,
               const float* __restrict__ bih_f, const float* __restrict__ bhh_f,
               const float* __restrict__ Wih_b, const float* __restrict__ Whh_b,
               const float* __restrict__ bih_b, const float* __restrict__ bhh_b,
               float* __restrict__ out_rnn)
{
    extern __shared__ unsigned char lds_raw[];
    unsigned short* sWih = (unsigned short*)lds_raw;
    unsigned short* sWhh = sWih + G_SZ * N_IN;
    unsigned short* sX   = sWhh + G_SZ * H_SZ;
    unsigned short* sH   = sX   + 2 * BT * N_IN;

    const int dir  = blockIdx.y;          // 0 = fwd, 1 = bwd
    const int bg   = blockIdx.x;          // batch group [0,16)
    const int tid  = threadIdx.x;
    const int wave = tid >> 5;            // [0,8): owns j-slice [16*wave, 16*wave+16)
    const int lane = tid & 31;
    const int lcol = lane & 15;           // A-row / B-column / D-column within tile
    const int half = lane >> 4;

    const float* Wih = dir ? Wih_b : Wih_f;
    const float* Whh = dir ? Whh_b : Whh_f;
    const float* bih = dir ? bih_b : bih_f;
    const float* bhh = dir ? bhh_b : bhh_f;

    // ---- one-time: weights -> LDS (bf16), zero h, stage x for step 0 ----
    for (int i = tid; i < G_SZ * N_IN;   i += 256) sWih[i] = f2bf(Wih[i]);
    for (int i = tid; i < G_SZ * H_SZ;   i += 256) sWhh[i] = f2bf(Whh[i]);
    for (int i = tid; i < 2 * BT * H_SZ; i += 256) sH[i]   = 0;

    const int r_st = tid >> 4;            // staging: row in tile
    const int c_st = (tid & 15) * 16;     // staging: 16 consecutive cols
    {
        const int t0 = dir ? (T_STEPS - 1) : 0;
        const float* src = y_aux + ((size_t)(bg * BT + r_st) * T_STEPS + t0) * N_IN + c_st;
        float xs[16];
        #pragma unroll
        for (int i = 0; i < 16; i += 4) {
            float4 v = *(const float4*)(src + i);
            xs[i] = v.x; xs[i + 1] = v.y; xs[i + 2] = v.z; xs[i + 3] = v.w;
        }
        uint4 u0, u1;
        u0.x = pack2bf(xs[0], xs[1]);  u0.y = pack2bf(xs[2], xs[3]);
        u0.z = pack2bf(xs[4], xs[5]);  u0.w = pack2bf(xs[6], xs[7]);
        u1.x = pack2bf(xs[8], xs[9]);  u1.y = pack2bf(xs[10], xs[11]);
        u1.z = pack2bf(xs[12], xs[13]); u1.w = pack2bf(xs[14], xs[15]);
        unsigned short* dst = sX + r_st * N_IN + c_st;
        *(uint4*)(dst)     = u0;
        *(uint4*)(dst + 8) = u1;
    }

    // per-lane gate biases for this wave's j-column
    float bi[3], bh[3];
    #pragma unroll
    for (int gi = 0; gi < 3; ++gi) {
        int g = gi * H_SZ + wave * 16 + lcol;
        bi[gi] = bih[g];
        bh[gi] = bhh[g];
    }

    // per-wave invariant fragment base pointers
    const unsigned short* wihB[3];
    const unsigned short* whhB[3];
    #pragma unroll
    for (int gi = 0; gi < 3; ++gi) {
        int g = gi * H_SZ + wave * 16 + lcol;
        wihB[gi] = sWih + g * N_IN + half * 16;
        whhB[gi] = sWhh + g * H_SZ + half * 16;
    }

    __syncthreads();

    v8f hreg = {};   // carried hidden state (f32, D-matrix layout)

    for (int s = 0; s < T_STEPS; ++s) {
        const int t   = dir ? (T_STEPS - 1 - s) : s;
        const int cur = s & 1;
        const int nxt = cur ^ 1;
        const bool do_stage = (s + 1) < T_STEPS;

        // issue next-step x global loads early (hide behind the WMMA pipeline)
        float xs[16];
        if (do_stage) {
            const int tn = dir ? (T_STEPS - 2 - s) : (s + 1);
            const float* src = y_aux + ((size_t)(bg * BT + r_st) * T_STEPS + tn) * N_IN + c_st;
            #pragma unroll
            for (int i = 0; i < 16; i += 4) {
                float4 v = *(const float4*)(src + i);
                xs[i] = v.x; xs[i + 1] = v.y; xs[i + 2] = v.z; xs[i + 3] = v.w;
            }
        }

        // ---- fused 12-stage WMMA pipeline: stages 0..7 x-proj, 8..11 h-proj ----
        const unsigned short* xA = sX + cur * BT * N_IN + lcol * N_IN + half * 16;
        const unsigned short* hA = sH + cur * BT * H_SZ + lcol * H_SZ + half * 16;

        v8f accx[3] = {v8f{}, v8f{}, v8f{}};
        v8f acch[3] = {v8f{}, v8f{}, v8f{}};

        FragB af[2];
        FragB bf[2][3];

        // preload stage 0 (x-proj kb=0)
        load_frag(af[0], xA, xA + 8);
        #pragma unroll
        for (int gi = 0; gi < 3; ++gi)
            load_frag(bf[0][gi], wihB[gi], wihB[gi] + 8);

        #pragma unroll
        for (int st = 0; st < 12; ++st) {
            const int p = st & 1;
            const int n = p ^ 1;
            // prefetch stage st+1
            if (st + 1 < 8) {
                const int kb = st + 1;
                load_frag(af[n], xA + kb * 32, xA + kb * 32 + 8);
                #pragma unroll
                for (int gi = 0; gi < 3; ++gi)
                    load_frag(bf[n][gi], wihB[gi] + kb * 32, wihB[gi] + kb * 32 + 8);
            } else if (st + 1 < 12) {
                const int kb = st + 1 - 8;
                load_frag(af[n], hA + kb * 32, hA + kb * 32 + 8);
                #pragma unroll
                for (int gi = 0; gi < 3; ++gi)
                    load_frag(bf[n][gi], whhB[gi] + kb * 32, whhB[gi] + kb * 32 + 8);
            }
            // consume stage st
            if (st < 8) {
                #pragma unroll
                for (int gi = 0; gi < 3; ++gi)
                    accx[gi] = wmma_bf16(af[p].v, bf[p][gi].v, accx[gi]);
            } else {
                #pragma unroll
                for (int gi = 0; gi < 3; ++gi)
                    acch[gi] = wmma_bf16(af[p].v, bf[p][gi].v, acch[gi]);
            }
        }

#if __has_builtin(__builtin_amdgcn_sched_group_barrier)
        // Pin the pipeline schedule: [DS_READ 16][WMMA 3], 10x [DS_READ 8][WMMA 3],
        // [WMMA 3] -> every WMMA group waits on only the oldest 8 of 16
        // in-flight ds_loads (s_wait_dscnt 0x8), never a full drain.
        __builtin_amdgcn_sched_group_barrier(0x100, 16, 0);  // DS read
        __builtin_amdgcn_sched_group_barrier(0x008, 3, 0);   // WMMA
        #pragma unroll
        for (int g = 0; g < 10; ++g) {
            __builtin_amdgcn_sched_group_barrier(0x100, 8, 0);
            __builtin_amdgcn_sched_group_barrier(0x008, 3, 0);
        }
        __builtin_amdgcn_sched_group_barrier(0x008, 3, 0);
#endif

        // ---- fused GRU gates (torch order r,z,n), f32 in registers ----
        #pragma unroll
        for (int i = 0; i < 8; ++i) {
            float r  = 1.f / (1.f + __expf(-(accx[0][i] + bi[0] + acch[0][i] + bh[0])));
            float z  = 1.f / (1.f + __expf(-(accx[1][i] + bi[1] + acch[1][i] + bh[1])));
            float nn = accx[2][i] + bi[2] + r * (acch[2][i] + bh[2]);
            float e2 = __expf(-2.f * nn);
            float th = (1.f - e2) / (1.f + e2);
            hreg[i]  = (1.f - z) * th + z * hreg[i];
        }

        // commit staged x tile for next step as 2x ds_store_b128
        if (do_stage) {
            uint4 u0, u1;
            u0.x = pack2bf(xs[0], xs[1]);  u0.y = pack2bf(xs[2], xs[3]);
            u0.z = pack2bf(xs[4], xs[5]);  u0.w = pack2bf(xs[6], xs[7]);
            u1.x = pack2bf(xs[8], xs[9]);  u1.y = pack2bf(xs[10], xs[11]);
            u1.z = pack2bf(xs[12], xs[13]); u1.w = pack2bf(xs[14], xs[15]);
            unsigned short* dst = sX + nxt * BT * N_IN + r_st * N_IN + c_st;
            *(uint4*)(dst)     = u0;
            *(uint4*)(dst + 8) = u1;
        }

        // publish h into the *other* sH buffer (bf16) and to global output (f32)
        {
            const int jg = wave * 16 + lcol;
            unsigned short* hd = sH + nxt * BT * H_SZ;
            #pragma unroll
            for (int i = 0; i < 8; ++i) {
                hd[(i + half * 8) * H_SZ + jg] = f2bf(hreg[i]);
            }
            float* op = out_rnn + ((size_t)t * B_SZ + bg * BT) * (2 * H_SZ) + dir * H_SZ + jg;
            #pragma unroll
            for (int i = 0; i < 8; ++i) {
                op[(size_t)(i + half * 8) * (2 * H_SZ)] = hreg[i];
            }
        }

        // single barrier per step: sX[nxt] + sH[nxt] committed, sX[cur]/sH[cur]
        // reads complete -> next step may read nxt and overwrite cur.
        __syncthreads();
    }
}

// ---- tiny FC head: temp = [zeros(B,L), hn0] ; mu/logvar = temp @ W^T + b ----
__global__ void __launch_bounds__(256)
fc_head_kernel(const float* __restrict__ rnn0,   // rnn_output[0] : [B, 2H]
               const float* __restrict__ Wmu, const float* __restrict__ bmu,
               const float* __restrict__ Wlv, const float* __restrict__ blv,
               float* __restrict__ mu, float* __restrict__ lv)
{
    const int idx = blockIdx.x * blockDim.x + threadIdx.x;
    if (idx >= B_SZ * L_SZ) return;
    const int b = idx / L_SZ;
    const int l = idx % L_SZ;
    const int F = 2 * H_SZ + L_SZ;   // 272; first L columns hit zeros
    float am = bmu[l];
    float av = blv[l];
    const float* hb = rnn0 + (size_t)b * (2 * H_SZ);
    const float* wm = Wmu + (size_t)l * F + L_SZ;
    const float* wl = Wlv + (size_t)l * F + L_SZ;
    for (int c = 0; c < 2 * H_SZ; c += 4) {
        float4 h = *(const float4*)(hb + c);
        float4 m = *(const float4*)(wm + c);
        float4 v = *(const float4*)(wl + c);
        am += h.x * m.x + h.y * m.y + h.z * m.z + h.w * m.w;
        av += h.x * v.x + h.y * v.y + h.z * v.z + h.w * v.w;
    }
    mu[idx] = am;
    lv[idx] = av;
}

extern "C" void kernel_launch(void* const* d_in, const int* in_sizes, int n_in,
                              void* d_out, int out_size, void* d_ws, size_t ws_size,
                              hipStream_t stream) {
    const float* y_aux   = (const float*)d_in[0];
    const float* Wih_f   = (const float*)d_in[1];
    const float* Whh_f   = (const float*)d_in[2];
    const float* bih_f   = (const float*)d_in[3];
    const float* bhh_f   = (const float*)d_in[4];
    const float* Wih_b   = (const float*)d_in[5];
    const float* Whh_b   = (const float*)d_in[6];
    const float* bih_b   = (const float*)d_in[7];
    const float* bhh_b   = (const float*)d_in[8];
    const float* W_mu    = (const float*)d_in[9];
    const float* b_mu    = (const float*)d_in[10];
    const float* W_lv    = (const float*)d_in[11];
    const float* b_lv    = (const float*)d_in[12];

    float* mu  = (float*)d_out;                    // [B, L]
    float* lv  = mu + B_SZ * L_SZ;                 // [B, L]
    float* rnn = lv + B_SZ * L_SZ;                 // [T, B, 2H]

    const size_t lds_bytes =
        (size_t)(G_SZ * N_IN + G_SZ * H_SZ + 2 * BT * N_IN + 2 * BT * H_SZ)
        * sizeof(unsigned short);

    dim3 grid(B_SZ / BT, 2);   // 16 batch groups x 2 directions = 32 persistent WGs
    gru_dir_kernel<<<grid, 256, lds_bytes, stream>>>(
        y_aux, Wih_f, Whh_f, bih_f, bhh_f, Wih_b, Whh_b, bih_b, bhh_b, rnn);

    fc_head_kernel<<<(B_SZ * L_SZ + 255) / 256, 256, 0, stream>>>(
        rnn, W_mu, b_mu, W_lv, b_lv, mu, lv);
}
